// LSTM_76553497084242
// MI455X (gfx1250) — matrix operand order repaired and checked by reading, not compiled, not run on hardware
//
#include <hip/hip_runtime.h>
#include <hip/hip_bf16.h>

// ---------------------------------------------------------------------------
// LSTM on MI455X (gfx1250):
//   Phase 1: XU = x @ U + bias  (big bf16 WMMA GEMM, M=32768,N=4096,K=1024)
//   Phase 2: persistent recurrent kernel:
//     - shared h A-tile staged in LDS (all 8 waves of a block share m)
//     - register double-buffered B tiles, 1-iteration prefetch distance
//     - XU epilogue operands prefetched at step start
//     - m-group scoped global barriers (agent release/acquire fences)
//     - ping-pong h buffers, c kept in registers for all 512 steps
// ---------------------------------------------------------------------------

typedef __bf16 bf16_t;
typedef bf16_t v16bf __attribute__((ext_vector_type(16)));
typedef bf16_t v8bf  __attribute__((ext_vector_type(8)));
typedef float  v8f   __attribute__((ext_vector_type(8)));
typedef float  f32x8 __attribute__((ext_vector_type(8)));

#define B_    64
#define S_    512
#define IN_   1024
#define HID_  1024
#define NG_   (4 * HID_)       // 4096 gate columns
#define M_    (B_ * S_)        // 32768 rows of x (2D view)
#define KT_   (IN_ / 32)       // 32 K-chunks of 32
#define LDSROW 1032            // padded row stride (halfs): 2064B -> rotate banks

__device__ __forceinline__ float fast_sigmoid(float x) {
  x = fminf(30.f, fmaxf(-30.f, x));
  return 1.f / (1.f + __expf(-x));
}
__device__ __forceinline__ float fast_tanh(float x) {
  x = fminf(15.f, fmaxf(-15.f, x));
  const float e = __expf(-2.f * x);
  return (1.f - e) / (1.f + e);
}

// ---------------------------------------------------------------------------
// Pack K x N (K=1024, N=4096) f32 weights into a WMMA B-operand layout
// (contiguous k order, same as the A loads; not the ISA table order): tile (nt,kt): lane holds 16 consecutive K of column
// nt*16+lane%16 starting at K = kt*32 + (lane&16). Packed [nt][kt][lane][16].
// ---------------------------------------------------------------------------
__global__ void __launch_bounds__(256)
pack_w_kernel(const float* __restrict__ src, bf16_t* __restrict__ dst) {
  int idx = blockIdx.x * 256 + threadIdx.x;          // total = 2^22
  if (idx >= IN_ * NG_) return;
  int e    = idx & 15;
  int lane = (idx >> 4) & 31;
  int kt   = (idx >> 9) & 31;
  int nt   = idx >> 14;
  int k    = kt * 32 + (lane & 16) + e;
  int col  = nt * 16 + (lane & 15);
  dst[idx] = (bf16_t)src[(long)k * NG_ + col];
}

__global__ void __launch_bounds__(256)
init_h_kernel(const float* __restrict__ h0, bf16_t* __restrict__ h, int n) {
  int i = blockIdx.x * 256 + threadIdx.x;
  if (i < n) h[i] = (bf16_t)h0[i];
}

// ---------------------------------------------------------------------------
// Phase 1 GEMM: XU[r][n] = sum_k x[r][k]*U[k][n] + Bias[n]   (f32 out)
// 131072 waves -> occupancy hides latency; leave scheduling to compiler.
// ---------------------------------------------------------------------------
__global__ void __launch_bounds__(256)
gemm_xu_kernel(const float*  __restrict__ X,     // [M_, IN_] f32
               const bf16_t* __restrict__ Upk,   // packed B operand
               const float*  __restrict__ Bias,  // [NG_]
               float*        __restrict__ XU)    // [M_, NG_]
{
  const int lane = threadIdx.x & 31;
  const int wid  = (blockIdx.x * 256 + threadIdx.x) >> 5;  // 0..131071
  const int mt   = wid >> 6;                                // 0..2047
  const int ngrp = wid & 63;                                // 0..63

  const int rowA = mt * 16 + (lane & 15);
  const int k0   = (lane >> 4) * 16;
  const float* xrow = X + (long)rowA * IN_ + k0;

  const v8f zero = {0.f,0.f,0.f,0.f,0.f,0.f,0.f,0.f};
  v8f acc[4];
  #pragma unroll
  for (int g = 0; g < 4; ++g) acc[g] = zero;

  #pragma unroll 4
  for (int kc = 0; kc < KT_; ++kc) {
    f32x8 lo = *(const f32x8*)(xrow + kc * 32);
    f32x8 hi = *(const f32x8*)(xrow + kc * 32 + 8);
    v16bf a;
    #pragma unroll
    for (int i = 0; i < 8; ++i) { a[i] = (bf16_t)lo[i]; a[8 + i] = (bf16_t)hi[i]; }
    #pragma unroll
    for (int g = 0; g < 4; ++g) {
      const int nt = ngrp * 4 + g;
      const v16bf b = *(const v16bf*)(Upk + (((nt * KT_ + kc) * 32 + lane) << 4));
      acc[g] = __builtin_amdgcn_wmma_f32_16x16x32_bf16(
                   false, a, false, b, (short)0, acc[g], false, false);
    }
  }

  const int col   = lane & 15;
  const int rbase = mt * 16 + ((lane >> 4) << 3);
  #pragma unroll
  for (int g = 0; g < 4; ++g) {
    const int colg = (ngrp * 4 + g) * 16 + col;
    const float bias = Bias[colg];
    #pragma unroll
    for (int v = 0; v < 8; ++v)
      XU[(long)(rbase + v) * NG_ + colg] = acc[g][v] + bias;
  }
}

// ---------------------------------------------------------------------------
// Phase 2: persistent recurrence. Grid = 32 WGs x 256 thr (8 waves).
// m = blockIdx>>3 (all waves of a block share m!), jt = (blockIdx&7)*8 + wave.
// ---------------------------------------------------------------------------
__global__ void __launch_bounds__(256)
lstm_seq_kernel(const bf16_t* __restrict__ Vpk,   // packed B operand of V
                const float*  __restrict__ XU,    // [B_,S_,NG_] bias folded
                const float*  __restrict__ c0,    // [B_,HID_]
                bf16_t*       __restrict__ hA,    // ping
                bf16_t*       __restrict__ hB,    // pong
                float*        __restrict__ out,   // [B_,S_,HID_]
                float*        __restrict__ hfin,  // [B_,HID_]
                float*        __restrict__ cfin,  // [B_,HID_]
                int*          __restrict__ counters) // [S_][4]
{
  __shared__ bf16_t smA[16 * LDSROW];         // 16 x 1024 h-tile, padded rows

  const int lane = threadIdx.x & 31;
  const int w    = threadIdx.x >> 5;          // wave in block 0..7
  const int m    = blockIdx.x >> 3;           // 0..3
  const int jt   = (blockIdx.x & 7) * 8 + w;  // 0..63

  const int ra   = lane & 15;                 // A row within tile
  const int k0   = (lane >> 4) * 16;          // A K-run select
  const int col  = lane & 15;
  const int rsub = (lane >> 4) << 3;          // 0 or 8
  const int xcol = jt * 16 + col;

  // per-gate packed-B base pointers (lane term folded in)
  const bf16_t* vb[4];
  #pragma unroll
  for (int g = 0; g < 4; ++g)
    vb[g] = Vpk + ((((g * 64 + jt) * KT_) * 32 + lane) << 4);

  // global h-tile bases for staging (rows m*16..m*16+15)
  const bf16_t* hbaseA = hA + (long)m * 16 * HID_;
  const bf16_t* hbaseB = hB + (long)m * 16 * HID_;

  // cell state lives in registers for the whole sequence
  float creg[8];
  #pragma unroll
  for (int v = 0; v < 8; ++v)
    creg[v] = c0[(m * 16 + rsub + v) * HID_ + xcol];

  const v8f zero = {0.f,0.f,0.f,0.f,0.f,0.f,0.f,0.f};

  for (int t = 0; t < S_; ++t) {
    // ---- prefetch XU operands for this step (hide HBM latency under WMMAs)
    float xu[4][8];
    #pragma unroll
    for (int v = 0; v < 8; ++v) {
      const long xb = ((long)(m * 16 + rsub + v) * S_ + t) * NG_ + xcol;
      #pragma unroll
      for (int g = 0; g < 4; ++g) xu[g][v] = XU[xb + g * HID_];
    }

    // ---- stage shared h tile (16 x 1024 bf16 = 32KB) into LDS, once/block
    {
      const bf16_t* hsrc = (t & 1) ? hbaseB : hbaseA;
      #pragma unroll
      for (int i = 0; i < 8; ++i) {
        const int c  = threadIdx.x + i * 256;   // 2048 chunks of 16B
        const int r  = c >> 7;
        const int kb = (c & 127) << 3;          // element offset (halfs)
        *(v8bf*)(&smA[r * LDSROW + kb]) = *(const v8bf*)(hsrc + r * HID_ + kb);
      }
    }
    __syncthreads();

    // ---- WMMA K-loop, register double-buffered (prefetch distance 1)
    v8f acc[4];
    #pragma unroll
    for (int g = 0; g < 4; ++g) acc[g] = zero;

    const bf16_t* arow = &smA[ra * LDSROW + k0];

    v16bf a_cur, b_cur[4];
    {
      v8bf lo = *(const v8bf*)(arow);
      v8bf hi = *(const v8bf*)(arow + 8);
      a_cur = __builtin_shufflevector(lo, hi, 0,1,2,3,4,5,6,7,8,9,10,11,12,13,14,15);
      #pragma unroll
      for (int g = 0; g < 4; ++g) b_cur[g] = *(const v16bf*)(vb[g]);
    }

    #pragma unroll
    for (int kc = 0; kc < KT_; ++kc) {
      v16bf a_nxt, b_nxt[4];
      if (kc + 1 < KT_) {
        v8bf lo = *(const v8bf*)(arow + (kc + 1) * 32);
        v8bf hi = *(const v8bf*)(arow + (kc + 1) * 32 + 8);
        a_nxt = __builtin_shufflevector(lo, hi, 0,1,2,3,4,5,6,7,8,9,10,11,12,13,14,15);
        #pragma unroll
        for (int g = 0; g < 4; ++g)
          b_nxt[g] = *(const v16bf*)(vb[g] + ((kc + 1) << 9));   // 32 lanes *16 halfs
      }
      #pragma unroll
      for (int g = 0; g < 4; ++g)
        acc[g] = __builtin_amdgcn_wmma_f32_16x16x32_bf16(
                     false, a_cur, false, b_cur[g], (short)0, acc[g], false, false);
      a_cur = a_nxt;
      #pragma unroll
      for (int g = 0; g < 4; ++g) b_cur[g] = b_nxt[g];
    }

    // ---- gate math + state update (each (row,col) uniquely owned)
    bf16_t* __restrict__ hw = ((t & 1) ? hA : hB);
    #pragma unroll
    for (int v = 0; v < 8; ++v) {
      const int  br = m * 16 + rsub + v;
      const float n  = fast_tanh   (acc[0][v] + xu[0][v]);
      const float ig = fast_sigmoid(acc[1][v] + xu[1][v]);
      const float fg = fast_sigmoid(acc[2][v] + xu[2][v]);
      const float og = fast_sigmoid(acc[3][v] + xu[3][v]);
      const float cn = creg[v] * fg + n * ig;
      creg[v] = cn;
      const float hn = fast_tanh(cn) * og;
      const int  ci  = br * HID_ + xcol;
      out[((long)br * S_ + t) * HID_ + xcol] = hn;
      hw[ci] = (bf16_t)hn;
      if (t == S_ - 1) { hfin[ci] = hn; cfin[ci] = cn; }
    }

    // ---- m-group barrier (8 workgroups) before next step's h reads
    if (t < S_ - 1) {
      __builtin_amdgcn_fence(__ATOMIC_RELEASE, "agent");   // publish h stores
      __syncthreads();                                      // also guards LDS reuse
      int* ctr = &counters[t * 4 + m];
      if (threadIdx.x == 0) {
        __hip_atomic_fetch_add(ctr, 1, __ATOMIC_RELAXED, __HIP_MEMORY_SCOPE_AGENT);
        while (__hip_atomic_load(ctr, __ATOMIC_RELAXED, __HIP_MEMORY_SCOPE_AGENT) < 8)
          __builtin_amdgcn_s_sleep(1);
      }
      __syncthreads();
      __builtin_amdgcn_fence(__ATOMIC_ACQUIRE, "agent");   // invalidate for h loads
    }
  }
}

// ---------------------------------------------------------------------------
extern "C" void kernel_launch(void* const* d_in, const int* in_sizes, int n_in,
                              void* d_out, int out_size, void* d_ws, size_t ws_size,
                              hipStream_t stream) {
  (void)in_sizes; (void)n_in; (void)out_size; (void)ws_size;

  const float* x    = (const float*)d_in[0];  // [B,S,IN]
  const float* U    = (const float*)d_in[1];  // [IN,4H]
  const float* V    = (const float*)d_in[2];  // [HID,4H]
  const float* Bias = (const float*)d_in[3];  // [4H]
  const float* h0   = (const float*)d_in[4];  // [B,HID]
  const float* c0   = (const float*)d_in[5];  // [B,HID]

  char*   ws   = (char*)d_ws;
  size_t  o    = 0;
  float*  XU   = (float*)(ws + o);  o += (size_t)M_ * NG_ * sizeof(float);   // 512 MiB
  bf16_t* Upk  = (bf16_t*)(ws + o); o += (size_t)IN_  * NG_ * sizeof(bf16_t);
  bf16_t* Vpk  = (bf16_t*)(ws + o); o += (size_t)HID_ * NG_ * sizeof(bf16_t);
  bf16_t* hA   = (bf16_t*)(ws + o); o += (size_t)B_ * HID_ * sizeof(bf16_t);
  bf16_t* hB   = (bf16_t*)(ws + o); o += (size_t)B_ * HID_ * sizeof(bf16_t);
  int*    ctrs = (int*)(ws + o);    o += (size_t)S_ * 4 * sizeof(int);

  float* out  = (float*)d_out;
  float* hfin = out + (size_t)B_ * S_ * HID_;
  float* cfin = hfin + (size_t)B_ * HID_;

  pack_w_kernel<<<(IN_ * NG_) / 256, 256, 0, stream>>>(U, Upk);
  pack_w_kernel<<<(HID_ * NG_) / 256, 256, 0, stream>>>(V, Vpk);
  init_h_kernel<<<(B_ * HID_ + 255) / 256, 256, 0, stream>>>(h0, hA, B_ * HID_);

  gemm_xu_kernel<<<(M_ / 16) * 64 / 8, 256, 0, stream>>>(x, Upk, Bias, XU);

  hipMemsetAsync(ctrs, 0, (size_t)S_ * 4 * sizeof(int), stream);
  lstm_seq_kernel<<<32, 256, 0, stream>>>(Vpk, XU, c0, hA, hB, out, hfin, cfin, ctrs);
}
